// RKMeansResidualQuantizer_54107997995398
// MI455X (gfx1250) — matrix operand order repaired and checked by reading, not compiled, not run on hardware
//
#include <hip/hip_runtime.h>
#include <math.h>

typedef __attribute__((ext_vector_type(16))) _Float16 v16h;
typedef __attribute__((ext_vector_type(8)))  float    v8f;
typedef __attribute__((ext_vector_type(4)))  float    v4f;

#define DIM   128
#define NCODE 1024
#define NLEV  4
#define BETA  0.25f

// ---------------------------------------------------------------------------
// Pre-pass: fp32 codebook -> f16 copy (for WMMA) + 0.5*|cb_k|^2 per code row.
// One block (128 threads) per (level,code) row.
// ---------------------------------------------------------------------------
__global__ __launch_bounds__(128) void rvq_prep(const float* __restrict__ cb,
                                                _Float16* __restrict__ cb16,
                                                float* __restrict__ hnorm) {
    const int rowk = blockIdx.x;              // 0 .. NLEV*NCODE-1
    const int t = threadIdx.x;                // 0 .. 127
    float v = cb[(size_t)rowk * DIM + t];
    cb16[(size_t)rowk * DIM + t] = (_Float16)v;
    float p = v * v;
    #pragma unroll
    for (int off = 16; off > 0; off >>= 1) p += __shfl_xor(p, off, 32);
    __shared__ float sp[4];
    if ((t & 31) == 0) sp[t >> 5] = p;
    __syncthreads();
    if (t == 0) hnorm[rowk] = 0.5f * (sp[0] + sp[1] + sp[2] + sp[3]);
}

// ---------------------------------------------------------------------------
// Main RVQ kernel. 128 threads = 4 waves; each wave owns 16 rows of x.
// Lane l owns row (l%16), dims {16q + 8*(l>=16) + s : q<8, s<8}  (64 fp32).
// This ownership equals the WMMA 16-bit A-matrix 16x32 fragment layout, so
// A fragments for all 4 K-chunks come straight from registers.
// ---------------------------------------------------------------------------
__global__ __launch_bounds__(128) void rvq_main(
    const float*    __restrict__ x,
    const float*    __restrict__ cb32,
    const _Float16* __restrict__ cb16,
    const float*    __restrict__ hnorm,
    float*          __restrict__ out_xq,
    float*          __restrict__ out_idx,
    float*          __restrict__ blockloss,
    int nblocks) {

    __shared__ int   s_idx[4][16];
    __shared__ float s_loss[4][NLEV];

    const int tid  = threadIdx.x;
    const int wave = tid >> 5;
    const int lane = tid & 31;
    const int half = lane >> 4;      // 0: dims s in [0,8); 1: s in [8,16)
    const int lid  = lane & 15;      // also: my row within the wave tile
    const int a    = half * 8;
    const long rowG = (long)blockIdx.x * 64 + wave * 16 + lid;

    float res[64];
    float xq[64];

    // Load my 64 dims of row rowG.
    const float* xrow = x + rowG * DIM;
    #pragma unroll
    for (int q = 0; q < 8; ++q) {
        v4f r0 = *(const v4f*)(xrow + q * 16 + a);
        v4f r1 = *(const v4f*)(xrow + q * 16 + a + 4);
        #pragma unroll
        for (int s = 0; s < 4; ++s) {
            res[q * 8 + s]     = r0[s];
            res[q * 8 + 4 + s] = r1[s];
            xq[q * 8 + s]      = 0.0f;
            xq[q * 8 + 4 + s]  = 0.0f;
        }
    }

    for (int lvl = 0; lvl < NLEV; ++lvl) {
        // residual = l2_normalize(residual) for levels > 0
        if (lvl > 0) {
            float sq = 0.0f;
            #pragma unroll
            for (int d = 0; d < 64; ++d) sq += res[d] * res[d];
            sq += __shfl_xor(sq, 16, 32);            // partner lane: other dim half
            float inv = 1.0f / fmaxf(sqrtf(sq), 1e-12f);
            #pragma unroll
            for (int d = 0; d < 64; ++d) res[d] *= inv;
        }

        // Build A fragments (16x32 f16) for the 4 K-chunks from registers.
        // Chunk c, half h<8  -> K = 32c +   a + h -> res[(2c)*8   + h]
        //          half h>=8 -> K = 32c+16+a+h-8 -> res[(2c+1)*8 + h-8]
        v16h af[4];
        #pragma unroll
        for (int c = 0; c < 4; ++c) {
            #pragma unroll
            for (int s = 0; s < 8; ++s) {
                af[c][s]     = (_Float16)res[(2 * c)     * 8 + s];
                af[c][s + 8] = (_Float16)res[(2 * c + 1) * 8 + s];
            }
        }

        float best[8];
        int   bidx[8];
        #pragma unroll
        for (int j = 0; j < 8; ++j) { best[j] = -__builtin_inff(); bidx[j] = 0; }

        // B fragment base: code k = t*16 + lid (lane%16 = column N), my K half.
        const _Float16* cbl = cb16 + ((size_t)lvl * NCODE + lid) * DIM + half * 8;
        const float*    hn  = hnorm + lvl * NCODE + lid;

        for (int t = 0; t < NCODE / 16; ++t) {
            const _Float16* bp = cbl + (size_t)t * 16 * DIM;
            v8f acc = {};
            #pragma unroll
            for (int c = 0; c < 4; ++c) {
                v16h bf;   // table order: s<8 -> K 8h+s, s>=8 -> K 16+8h+s-8
                #pragma unroll
                for (int s = 0; s < 8; ++s) { bf[s] = bp[c * 32 + s]; bf[s + 8] = bp[c * 32 + 16 + s]; }
                acc = __builtin_amdgcn_wmma_f32_16x16x32_f16(
                    false, af[c], false, bf, (short)0, acc, false, false);
            }
            // score = r.cb - 0.5|cb|^2 ; argmax == argmin of squared distance
            float bias = hn[t * 16];
            int kc = t * 16 + lid;
            #pragma unroll
            for (int j = 0; j < 8; ++j) {
                float sc = acc[j] - bias;                // slot j <-> row (half?8:0)+j
                if (sc > best[j]) { best[j] = sc; bidx[j] = kc; }
            }
        }

        // Butterfly argmax across the 16 lanes of each half (rows partitioned
        // by half exactly as the C/D matrix layout). Tie -> smaller index.
        #pragma unroll
        for (int off = 1; off <= 8; off <<= 1) {
            #pragma unroll
            for (int j = 0; j < 8; ++j) {
                float os = __shfl_xor(best[j], off, 32);
                int   oi = __shfl_xor(bidx[j], off, 32);
                if (os > best[j] || (os == best[j] && oi < bidx[j])) {
                    best[j] = os; bidx[j] = oi;
                }
            }
        }

        if (lid == 0) {
            #pragma unroll
            for (int j = 0; j < 8; ++j) {
                int r = half * 8 + j;
                s_idx[wave][r] = bidx[j];
                long rg = (long)blockIdx.x * 64 + wave * 16 + r;
                out_idx[rg * NLEV + lvl] = (float)bidx[j];
            }
        }
        __syncthreads();

        // Gather fp32 codeword, update residual / x_q / loss.
        int k = s_idx[wave][lid];
        const float* crow = cb32 + ((size_t)lvl * NCODE + k) * DIM;
        float lsum = 0.0f;
        #pragma unroll
        for (int q = 0; q < 8; ++q) {
            v4f c0 = *(const v4f*)(crow + q * 16 + a);
            v4f c1 = *(const v4f*)(crow + q * 16 + a + 4);
            #pragma unroll
            for (int s = 0; s < 4; ++s) {
                float e0 = res[q * 8 + s]     - c0[s];
                float e1 = res[q * 8 + 4 + s] - c1[s];
                lsum += e0 * e0 + e1 * e1;
                xq[q * 8 + s]     += c0[s];
                xq[q * 8 + 4 + s] += c1[s];
                res[q * 8 + s]     = e0;
                res[q * 8 + 4 + s] = e1;
            }
        }
        #pragma unroll
        for (int off = 16; off > 0; off >>= 1) lsum += __shfl_xor(lsum, off, 32);
        if (lane == 0) s_loss[wave][lvl] = lsum;
        __syncthreads();
    }

    // Write x_q.
    float* orow = out_xq + rowG * DIM;
    #pragma unroll
    for (int q = 0; q < 8; ++q) {
        v4f o0, o1;
        #pragma unroll
        for (int s = 0; s < 4; ++s) { o0[s] = xq[q * 8 + s]; o1[s] = xq[q * 8 + 4 + s]; }
        *(v4f*)(orow + q * 16 + a)     = o0;
        *(v4f*)(orow + q * 16 + a + 4) = o1;
    }

    // Per-block deterministic loss partials (fixed wave order).
    if (tid < NLEV) {
        float s = 0.0f;
        #pragma unroll
        for (int w = 0; w < 4; ++w) s += s_loss[w][tid];
        blockloss[(size_t)tid * nblocks + blockIdx.x] = s;
    }
}

// ---------------------------------------------------------------------------
// Deterministic final reduction: mean_loss = (1+BETA)/(NLEV*N*DIM) * sum(all)
// ---------------------------------------------------------------------------
__global__ __launch_bounds__(256) void rvq_loss_final(const float* __restrict__ blockloss,
                                                      int count, float scale,
                                                      float* __restrict__ out) {
    __shared__ float s[256];
    int t = threadIdx.x;
    float acc = 0.0f;
    for (int i = t; i < count; i += 256) acc += blockloss[i];
    s[t] = acc;
    __syncthreads();
    for (int off = 128; off > 0; off >>= 1) {
        if (t < off) s[t] += s[t + off];
        __syncthreads();
    }
    if (t == 0) out[0] = s[0] * scale;
}

extern "C" void kernel_launch(void* const* d_in, const int* in_sizes, int n_in,
                              void* d_out, int out_size, void* d_ws, size_t ws_size,
                              hipStream_t stream) {
    const float* x    = (const float*)d_in[0];   // [N, 128]
    const float* cb32 = (const float*)d_in[1];   // [4, 1024, 128]
    const long N       = (long)in_sizes[0] / DIM;      // 131072
    const int  nblocks = (int)(N / 64);                // 64 rows per block
    const size_t cb_elems = (size_t)in_sizes[1];       // 4*1024*128

    // Workspace layout: f16 codebook | half-norms | per-block loss partials
    char* ws = (char*)d_ws;
    _Float16* cb16  = (_Float16*)ws;
    float*    hnorm = (float*)(ws + cb_elems * sizeof(_Float16));
    float*    bloss = (float*)(ws + cb_elems * sizeof(_Float16)
                                  + (cb_elems / DIM) * sizeof(float));

    float* out_xq   = (float*)d_out;                   // [N,128]
    float* out_loss = out_xq + N * DIM;                // scalar
    float* out_idx  = out_loss + 1;                    // [N,4] as floats

    rvq_prep<<<(int)(cb_elems / DIM), 128, 0, stream>>>(cb32, cb16, hnorm);
    rvq_main<<<nblocks, 128, 0, stream>>>(x, cb32, cb16, hnorm,
                                          out_xq, out_idx, bloss, nblocks);
    const float scale = (1.0f + BETA) / ((float)NLEV * (float)N * (float)DIM);
    rvq_loss_final<<<1, 256, 0, stream>>>(bloss, NLEV * nblocks, scale, out_loss);
}
